// Transducer_75797582840102
// MI455X (gfx1250) — matrix operand order repaired and checked by reading, not compiled, not run on hardware
//
#include <hip/hip_runtime.h>

#define B_     4
#define T_     200
#define U_     100
#define U1_    101
#define DENC_  512
#define INNER_ 1024
#define VOC_   256

static constexpr float NEGF = -1e30f;

typedef __attribute__((ext_vector_type(16))) __bf16 v16bf;
typedef __attribute__((ext_vector_type(8)))  float  v8f;

static __device__ __forceinline__ unsigned short f2bf(float f) {
    unsigned int u = __float_as_uint(f);
    u += 0x7fffu + ((u >> 16) & 1u);           // round-to-nearest-even
    return (unsigned short)(u >> 16);
}
static __device__ __forceinline__ __bf16 us2bf(unsigned short u) {
    return __builtin_bit_cast(__bf16, u);
}

// ---------------------------------------------------------------- conversions
__global__ void cvt_bf16_kernel(const float* __restrict__ in,
                                unsigned short* __restrict__ out, int n) {
    int i = blockIdx.x * blockDim.x + threadIdx.x;
    if (i < n) out[i] = f2bf(in[i]);
}

// ------------------------------------------------- enc_p / dec_p projections
// out[row][n] = sum_k A[row][k] * W[n][koff + k],  K = 512, N = 1024
__global__ __launch_bounds__(256) void proj_gemm_kernel(
    const unsigned short* __restrict__ Abf,   // [rows][512] bf16
    const unsigned short* __restrict__ Wbf,   // [1024][1024] bf16 (W_fwd)
    int rows, int koff, float* __restrict__ out /* [rows][1024] */) {
    __shared__ unsigned short As[16][520];    // padded rows vs bank conflicts
    const int tid = threadIdx.x;
    const int r0  = blockIdx.x * 16;
    for (int idx = tid; idx < 16 * DENC_; idx += 256) {
        int m = idx >> 9, k = idx & (DENC_ - 1);
        int row = r0 + m;
        As[m][k] = (row < rows) ? Abf[row * DENC_ + k] : (unsigned short)0;
    }
    __syncthreads();

    const int wave = tid >> 5, lane = tid & 31;
    const int mrow = lane & 15, hi = lane >> 4;
    const int n0 = (blockIdx.y * 8 + wave) * 16;
    const int n  = n0 + (lane & 15);
    v8f acc = {};
    for (int kc = 0; kc < DENC_ / 32; ++kc) {
        // A fragment: lane holds row m, 16 contiguous K values
        int ka = kc * 32 + (hi << 4);
        v16bf a;
        #pragma unroll
        for (int j = 0; j < 8; ++j) {
            a[j]     = us2bf(As[mrow][ka + j]);
            a[j + 8] = us2bf(As[mrow][ka + 8 + j]);
        }
        // B fragment: lane holds col n, 16 contiguous K values
        const unsigned short* wp = Wbf + n * INNER_ + koff + kc * 32 + (hi << 4);
        v16bf bv;
        #pragma unroll
        for (int j = 0; j < 16; ++j) bv[j] = us2bf(wp[j]);
        acc = __builtin_amdgcn_wmma_f32_16x16x32_bf16(false, a, false, bv,
                                                      (short)0, acc, false, false);
    }
    #pragma unroll
    for (int r = 0; r < 8; ++r) {
        int M = (hi << 3) + r;
        int row = r0 + M;
        if (row < rows) out[row * INNER_ + n] = acc[r];
    }
}

// -------------------------------- fused joint: tanh + GEMM(N=256,K=1024) + LSE
__global__ __launch_bounds__(256) void joint_kernel(
    const float* __restrict__ enc_p, const float* __restrict__ dec_p,
    const float* __restrict__ b_fwd, const unsigned short* __restrict__ Wpbf,
    const float* __restrict__ b_proj, const int* __restrict__ targets,
    float* __restrict__ lpb, float* __restrict__ lpy) {
    __shared__ unsigned short As[16][1032];   // 16 x 1024 bf16 A-tile (padded)
    __shared__ float Ls[16][264];             // 16 x 256 logits (padded)

    const int tid = threadIdx.x;
    const int b = blockIdx.z, u = blockIdx.y, t0 = blockIdx.x * 16;

    const float* ep = enc_p + (size_t)(b * T_ + t0) * INNER_;
    const float* dp = dec_p + (size_t)(b * U1_ + u) * INNER_;
    for (int idx = tid; idx < 16 * INNER_; idx += 256) {
        int m = idx >> 10, k = idx & (INNER_ - 1);
        float v = 0.f;
        if (t0 + m < T_) v = tanhf(ep[m * INNER_ + k] + dp[k] + b_fwd[k]);
        As[m][k] = f2bf(v);
    }
    __syncthreads();

    const int wave = tid >> 5, lane = tid & 31;
    const int mrow = lane & 15, hi = lane >> 4, nc = lane & 15;
    const int n0a = (wave * 2 + 0) * 16, n0b = (wave * 2 + 1) * 16;
    v8f acc0 = {}, acc1 = {};
    for (int kc = 0; kc < INNER_ / 32; ++kc) {
        int ka = kc * 32 + (hi << 4);
        v16bf a;
        #pragma unroll
        for (int j = 0; j < 8; ++j) {
            a[j]     = us2bf(As[mrow][ka + j]);
            a[j + 8] = us2bf(As[mrow][ka + 8 + j]);
        }
        int kb = kc * 32 + (hi << 4);
        const unsigned short* w0 = Wpbf + (n0a + nc) * INNER_ + kb;
        const unsigned short* w1 = Wpbf + (n0b + nc) * INNER_ + kb;
        __builtin_prefetch(w0 + 32, 0, 1);    // next k-chunk of W_proj
        __builtin_prefetch(w1 + 32, 0, 1);
        v16bf b0, b1;
        #pragma unroll
        for (int j = 0; j < 16; ++j) { b0[j] = us2bf(w0[j]); b1[j] = us2bf(w1[j]); }
        acc0 = __builtin_amdgcn_wmma_f32_16x16x32_bf16(false, a, false, b0,
                                                       (short)0, acc0, false, false);
        acc1 = __builtin_amdgcn_wmma_f32_16x16x32_bf16(false, a, false, b1,
                                                       (short)0, acc1, false, false);
    }
    #pragma unroll
    for (int r = 0; r < 8; ++r) {
        int M = (hi << 3) + r;
        Ls[M][n0a + nc] = acc0[r] + b_proj[n0a + nc];
        Ls[M][n0b + nc] = acc1[r] + b_proj[n0b + nc];
    }
    __syncthreads();

    // log-softmax pieces: 16 threads per row reduce 256 vocab entries
    const int r = tid >> 4, c0 = tid & 15;
    float mx = -3.4e38f;
    #pragma unroll
    for (int i = 0; i < 16; ++i) mx = fmaxf(mx, Ls[r][c0 + 16 * i]);
    #pragma unroll
    for (int off = 8; off >= 1; off >>= 1) mx = fmaxf(mx, __shfl_xor(mx, off, 32));
    float s = 0.f;
    #pragma unroll
    for (int i = 0; i < 16; ++i) s += __expf(Ls[r][c0 + 16 * i] - mx);
    #pragma unroll
    for (int off = 8; off >= 1; off >>= 1) s += __shfl_xor(s, off, 32);
    if (c0 == 0) {
        int t = t0 + r;
        if (t < T_) {
            float lse = mx + __logf(s);
            lpb[(size_t)(b * T_ + t) * U1_ + u] = Ls[r][0] - lse;
            if (u < U_) {
                int tg = targets[b * U_ + u];
                lpy[(size_t)(b * T_ + t) * U_ + u] = Ls[r][tg] - lse;
            }
        }
    }
}

// ----------------------------------------- wavefront RNN-T alpha recursion
__global__ __launch_bounds__(128) void alpha_kernel(
    const float* __restrict__ lpb, const float* __restrict__ lpy,
    const int* __restrict__ ilen, const int* __restrict__ tlen,
    float* __restrict__ loss_part) {
    __shared__ float diag[2][U1_];
    __shared__ float result;
    const int b = blockIdx.x, u = threadIdx.x;
    const int TL = tlen[b];
    const int ti = ilen[b] - 1, ui = TL;
    const float* lpb_b = lpb + (size_t)b * T_ * U1_;
    const float* lpy_b = lpy + (size_t)b * T_ * U_;

    for (int d = 0; d < T_ + U_; ++d) {
        float* cur        = diag[d & 1];
        const float* prev = diag[(d ^ 1) & 1];
        int t = d - u;
        if (u <= U_ && t >= 0 && t < T_) {
            float v;
            if (t == 0 && u == 0) {
                v = 0.f;
            } else {
                float bt = NEGF, et = NEGF;
                if (t > 0) bt = prev[u] + lpb_b[(t - 1) * U1_ + u];
                if (u > 0) {
                    float ly = (u - 1 < TL) ? lpy_b[t * U_ + (u - 1)] : NEGF;
                    et = prev[u - 1] + ly;
                }
                float m = fmaxf(bt, et), n = fminf(bt, et);
                v = (m <= -1e29f) ? NEGF : m + log1pf(__expf(n - m));
            }
            cur[u] = v;
            if (t == ti && u == ui) result = v + lpb_b[ti * U1_ + ui];
        }
        __syncthreads();
    }
    if (threadIdx.x == 0) loss_part[b] = -result;
}

__global__ void finalize_kernel(const float* __restrict__ loss_part,
                                float* __restrict__ out) {
    if (threadIdx.x == 0 && blockIdx.x == 0) {
        float s = 0.f;
        for (int i = 0; i < B_; ++i) s += loss_part[i];
        out[0] = s / (float)B_;
    }
}

// ------------------------------------------------------------------- launcher
extern "C" void kernel_launch(void* const* d_in, const int* in_sizes, int n_in,
                              void* d_out, int out_size, void* d_ws, size_t ws_size,
                              hipStream_t stream) {
    const float* enc  = (const float*)d_in[0];
    const float* dec  = (const float*)d_in[1];
    const float* Wf   = (const float*)d_in[2];
    const float* bfwd = (const float*)d_in[3];
    const float* Wp   = (const float*)d_in[4];
    const float* bp   = (const float*)d_in[5];
    const int*   tgt  = (const int*)d_in[6];
    const int*   il   = (const int*)d_in[7];
    const int*   tl   = (const int*)d_in[8];
    float* out = (float*)d_out;

    char* base = (char*)d_ws;
    size_t off = 0;
    auto take = [&](size_t bytes) -> char* {
        char* p = base + off;
        off += (bytes + 255) & ~(size_t)255;
        return p;
    };
    unsigned short* Wf_bf  = (unsigned short*)take((size_t)INNER_ * INNER_ * 2);
    unsigned short* Wp_bf  = (unsigned short*)take((size_t)VOC_ * INNER_ * 2);
    unsigned short* enc_bf = (unsigned short*)take((size_t)B_ * T_ * DENC_ * 2);
    unsigned short* dec_bf = (unsigned short*)take((size_t)B_ * U1_ * DENC_ * 2);
    float* enc_p = (float*)take((size_t)B_ * T_ * INNER_ * 4);
    float* dec_p = (float*)take((size_t)B_ * U1_ * INNER_ * 4);
    float* lpb   = (float*)take((size_t)B_ * T_ * U1_ * 4);
    float* lpy   = (float*)take((size_t)B_ * T_ * U_ * 4);
    float* lpart = (float*)take(B_ * 4);

    auto cvt = [&](const float* i, unsigned short* o, int n) {
        cvt_bf16_kernel<<<(n + 255) / 256, 256, 0, stream>>>(i, o, n);
    };
    cvt(Wf, Wf_bf, INNER_ * INNER_);
    cvt(Wp, Wp_bf, VOC_ * INNER_);
    cvt(enc, enc_bf, B_ * T_ * DENC_);
    cvt(dec, dec_bf, B_ * U1_ * DENC_);

    proj_gemm_kernel<<<dim3((B_ * T_ + 15) / 16, 8), 256, 0, stream>>>(
        enc_bf, Wf_bf, B_ * T_, 0, enc_p);
    proj_gemm_kernel<<<dim3((B_ * U1_ + 15) / 16, 8), 256, 0, stream>>>(
        dec_bf, Wf_bf, B_ * U1_, DENC_, dec_p);

    joint_kernel<<<dim3((T_ + 15) / 16, U1_, B_), 256, 0, stream>>>(
        enc_p, dec_p, bfwd, Wp_bf, bp, tgt, lpb, lpy);

    alpha_kernel<<<B_, 128, 0, stream>>>(lpb, lpy, il, tl, lpart);
    finalize_kernel<<<1, 32, 0, stream>>>(lpart, out);

    (void)in_sizes; (void)n_in; (void)out_size; (void)ws_size;
}
